// FishEyePatchEmbedding_61074434949627
// MI455X (gfx1250) — matrix operand order repaired and checked by reading, not compiled, not run on hardware
//
#include <hip/hip_runtime.h>
#include <hip/hip_bf16.h>

// ---- problem constants -----------------------------------------------------
#define IMG_PIX   541696u     // 736*736
#define PLEN      768         // patch_length (angle samples per ring)
#define EMBED     768         // output channels
#define KDIM      768         // 3*16*16
#define PPB       1104        // patches per batch = 23*48
#define MTOT      35328       // 32*1104
#define MTILE     64
#define NTILE     128
#define KCHUNK    32
#define LDS_STR   40          // halves per LDS row (32 data + 8 pad -> 80B)
#define BUF_HALVES (MTILE * LDS_STR)   // 2560 halves per buffer

typedef __attribute__((ext_vector_type(16))) __bf16        v16bf;
typedef __attribute__((ext_vector_type(8)))  __bf16        v8bf;
typedef __attribute__((ext_vector_type(8)))  float         v8f;

union AFrag { v16bf v; v8bf h[2]; };

// ---- pass 1: conv_w (OC x K, f32) -> Wt (K x N, bf16) ----------------------
__global__ void fisheye_wt_kernel(const float* __restrict__ cw,
                                  __bf16* __restrict__ Wt) {
  unsigned k = blockIdx.x;                               // 0..767
  for (unsigned n = threadIdx.x; n < EMBED; n += 256)
    Wt[k * EMBED + n] = (__bf16)cw[n * KDIM + k];        // RNE fptrunc
}

// ---- pass 2: fused gather+lerp + patch-embed GEMM (bf16 WMMA, f32 acc) -----
__global__ void __launch_bounds__(256)
fisheye_gemm_kernel(const float* __restrict__ x,
                    const int*   __restrict__ idx0,
                    const int*   __restrict__ idx1,
                    const float* __restrict__ wgt,
                    const __bf16* __restrict__ Wt,
                    const float* __restrict__ bias,
                    float* __restrict__ out) {
  __shared__ __bf16 As[2 * BUF_HALVES];                  // 2 x 5120 B

  const int tid  = threadIdx.x;
  const int wave = tid >> 5;
  const int lane = tid & 31;
  const int lrow = lane & 15;
  const int lhi  = lane >> 4;

  const int m0 = blockIdx.x * MTILE;
  const int n0 = blockIdx.y * NTILE;

  // --- per-thread A-fill coordinates (depend only on tid/block) ---
  const int frow  = tid >> 2;                 // 0..63   (LDS/A row)
  const int kbase = (tid & 3) << 3;           // 0,8,16,24
  const int m_f   = m0 + frow;
  const unsigned b_f = (unsigned)m_f / (unsigned)PPB;
  const int p_f   = m_f - (int)b_f * PPB;
  const unsigned oy = (unsigned)p_f / 48u;
  const int ox    = p_f - (int)oy * 48;
  const unsigned rowq0 = ((unsigned)oy * 16u) * PLEN + (unsigned)(ox * 16);
  const unsigned xbat  = b_f * 3u * IMG_PIX;  // batch base (element offset)

  const int4*   I0 = (const int4*)idx0;
  const int4*   I1 = (const int4*)idx1;
  const float4* WF = (const float4*)wgt;

  // fill LDS A buffer at half-offset boff for K-chunk starting at kk
  auto fillA = [&](unsigned boff, int kk) {
    const int k0  = kk + kbase;               // 8 consecutive k, same (c,py)
    const unsigned c  = (unsigned)k0 >> 8;
    const unsigned py = ((unsigned)k0 >> 4) & 15u;
    const unsigned px0 = (unsigned)k0 & 15u;  // 0 or 8
    const unsigned g0 = rowq0 + py * PLEN + px0;        // multiple of 8
    const unsigned gq = g0 >> 2;
    const unsigned xoff = xbat + c * IMG_PIX;

    const int4 a0 = I0[gq], a1 = I0[gq + 1];
    const int4 c0 = I1[gq], c1 = I1[gq + 1];
    const float4 w0 = WF[gq], w1 = WF[gq + 1];

    const unsigned ia[8] = {(unsigned)a0.x, (unsigned)a0.y, (unsigned)a0.z, (unsigned)a0.w,
                            (unsigned)a1.x, (unsigned)a1.y, (unsigned)a1.z, (unsigned)a1.w};
    const unsigned ib[8] = {(unsigned)c0.x, (unsigned)c0.y, (unsigned)c0.z, (unsigned)c0.w,
                            (unsigned)c1.x, (unsigned)c1.y, (unsigned)c1.z, (unsigned)c1.w};
    const float    wf[8] = {w0.x, w0.y, w0.z, w0.w, w1.x, w1.y, w1.z, w1.w};

    v8bf pk;
    #pragma unroll
    for (int u = 0; u < 8; ++u) {
      float v0 = x[xoff + ia[u]];             // saddr + 32-bit voffset
      float v1 = x[xoff + ib[u]];
      pk[u] = (__bf16)(v0 + (v1 - v0) * wf[u]);   // -> v_cvt_pk_bf16_f32
    }
    *(v8bf*)(As + boff + (unsigned)(frow * LDS_STR + kbase)) = pk;  // 16B ds_store
  };

  v8f acc[2][2];
  #pragma unroll
  for (int i = 0; i < 2; ++i)
    #pragma unroll
    for (int j = 0; j < 2; ++j) acc[i][j] = (v8f){};

  const int mw = (wave & 1) * 32;             // wave M offset within tile
  const int nw = (wave >> 1) * 32;            // wave N offset within tile
  const unsigned ncol[2] = {(unsigned)(n0 + nw), (unsigned)(n0 + nw + 16)};

  fillA(0u, 0);

  for (int kk = 0, it = 0; kk < KDIM; kk += KCHUNK, ++it) {
    __syncthreads();
    // prefetch next chunk into the other buffer while this one is consumed
    if (kk + KCHUNK < KDIM)
      fillA((unsigned)(((it + 1) & 1) * BUF_HALVES), kk + KCHUNK);

    const __bf16* Ah = As + (unsigned)((it & 1) * BUF_HALVES);

    AFrag fa[2];
    #pragma unroll
    for (int ms = 0; ms < 2; ++ms) {
      const int arow = mw + ms * 16 + lrow;
      const int base = arow * LDS_STR + lhi * 8;
      fa[ms].h[0] = *(const v8bf*)(Ah + base);          // K{0..7} / {8..15}
      fa[ms].h[1] = *(const v8bf*)(Ah + base + 16);     // K{16..23} / {24..31}
    }
    AFrag fb[2];
    #pragma unroll
    for (int ns = 0; ns < 2; ++ns) {
      #pragma unroll
      for (int s = 0; s < 16; ++s) {
        const int kl = (s < 8) ? (lhi * 8 + s) : (16 + lhi * 8 + (s - 8));
        fb[ns].v[s] = Wt[(unsigned)(kk + kl) * (unsigned)EMBED + ncol[ns] + (unsigned)lrow];
      }
    }

    #pragma unroll
    for (int ms = 0; ms < 2; ++ms)
      #pragma unroll
      for (int ns = 0; ns < 2; ++ns)
        acc[ms][ns] = __builtin_amdgcn_wmma_f32_16x16x32_bf16(
            false, fa[ms].v, false, fb[ns].v,
            (short)0, acc[ms][ns], false, false);
  }

  // ---- epilogue: C layout -> NCHW output + bias ----
  #pragma unroll
  for (int ns = 0; ns < 2; ++ns) {
    const unsigned n = ncol[ns] + (unsigned)lrow;
    const float bv = bias[n];
    #pragma unroll
    for (int ms = 0; ms < 2; ++ms) {
      const int mbase = m0 + mw + ms * 16 + lhi * 8;
      #pragma unroll
      for (int j = 0; j < 8; ++j) {
        const int m = mbase + j;
        const unsigned b = (unsigned)m / (unsigned)PPB;
        const unsigned p = (unsigned)m - b * (unsigned)PPB;
        out[(b * (unsigned)EMBED + n) * (unsigned)PPB + p] = acc[ms][ns][j] + bv;
      }
    }
  }
}

// ---- host launcher ---------------------------------------------------------
extern "C" void kernel_launch(void* const* d_in, const int* in_sizes, int n_in,
                              void* d_out, int out_size, void* d_ws, size_t ws_size,
                              hipStream_t stream) {
  (void)in_sizes; (void)n_in; (void)out_size; (void)ws_size;
  const float* x      = (const float*)d_in[0];
  const float* conv_w = (const float*)d_in[1];
  const float* conv_b = (const float*)d_in[2];
  const int*   idx0   = (const int*)d_in[3];
  const int*   idx1   = (const int*)d_in[4];
  const float* w      = (const float*)d_in[5];
  float* out = (float*)d_out;
  __bf16* Wt = (__bf16*)d_ws;                   // 768*768*2 B = 1.18 MB

  fisheye_wt_kernel<<<dim3(KDIM), dim3(256), 0, stream>>>(conv_w, Wt);

  dim3 grid(MTOT / MTILE, EMBED / NTILE);       // 552 x 6
  fisheye_gemm_kernel<<<grid, dim3(256), 0, stream>>>(
      x, idx0, idx1, w, Wt, conv_b, out);
}
